// AdvancedTransformer_3478923510333
// MI455X (gfx1250) — matrix operand order repaired and checked by reading, not compiled, not run on hardware
//
#include <hip/hip_runtime.h>
#include <hip/hip_bf16.h>

// ---------------------------------------------------------------------------
// MI455X (gfx1250) fused "episodic/semantic memory" kernel.
//   episodic = softmax(x@We^T + be) @ ep_mem     (flash-style, split over slots)
//   semantic = softmax(x@Ws^T + bs) @ sem_mem    (flash-style, split over slots)
//   out      = [episodic, x] @ Wc^T + bc
// Precision: weights/activations converted fp32 -> bf16 on the fly (v_perm),
// accumulation in fp32 via v_wmma_f32_16x16x32_bf16 (wave32).
// V (memory bank) matrices are re-staged through LDS in a pair-packed bf16
// layout so all global traffic is coalesced b128 and the K-strided gather
// happens in LDS (conflict-free ds_load_b32).
// ---------------------------------------------------------------------------

typedef __bf16 v16bf __attribute__((ext_vector_type(16)));
typedef float  v8f   __attribute__((ext_vector_type(8)));

union BF16x16 { v16bf v; unsigned int u[8]; };

#define HD 1024   // hidden size
#define MT 16     // rows per M tile
#define NT 128    // slots per flash tile (8 waves x 16)
#define VSTR 1028 // vstage pair-row stride in dwords (4-bank skew)
#define PART_STRIDE 16416   // 16*1024 O + 16 m + 16 s (floats)

// pack two fp32 -> two bf16 (truncation) with a single v_perm_b32
__device__ __forceinline__ unsigned int pack_bf16(float lo, float hi) {
  return __builtin_amdgcn_perm(__float_as_uint(hi), __float_as_uint(lo), 0x07060302u);
}

// Stage a 16x1024 fp32 tile (row-major, ld=1024) into LDS as bf16, stride 1032.
__device__ __forceinline__ void stage16x1024(const float* __restrict__ src,
                                             unsigned short* lds, int tid) {
#pragma unroll
  for (int i = 0; i < 16; ++i) {
    int idx = tid + i * 256;      // float4 index, 4096 total
    int row = idx >> 8;           // 256 float4 per row
    int c4  = idx & 255;
    float4 f = *(const float4*)(src + (size_t)row * HD + c4 * 4);
    unsigned int p0 = pack_bf16(f.x, f.y);
    unsigned int p1 = pack_bf16(f.z, f.w);
    *(uint2*)(lds + row * 1032 + c4 * 4) = make_uint2(p0, p1);
  }
}

// Stage 32 V-rows (slots) x 1024 cols into LDS, pair-packed:
// vs[pr*VSTR + col] = bf16pair(V[slot0+2pr][col], V[slot0+2pr+1][col]).
// Global side: coalesced float4 loads; LDS side: uint4 stores.
__device__ __forceinline__ void stageV(const float* __restrict__ Vm, int slot0,
                                       unsigned int* vs, int tid) {
#pragma unroll
  for (int i = 0; i < 16; ++i) {
    int idx = tid + i * 256;      // uint4 index, 4096 total (16 pair-rows x 256)
    int pr  = idx >> 8;
    int c4  = (idx & 255) * 4;
    const float* p0 = Vm + (size_t)(slot0 + 2 * pr) * HD + c4;
    float4 a = *(const float4*)(p0);
    float4 b = *(const float4*)(p0 + HD);
    uint4 o;
    o.x = pack_bf16(a.x, b.x);
    o.y = pack_bf16(a.y, b.y);
    o.z = pack_bf16(a.z, b.z);
    o.w = pack_bf16(a.w, b.w);
    *(uint4*)(vs + pr * VSTR + c4) = o;
  }
}

// A fragment (16xK bf16, 32 K-slice at kb) from LDS.
// Lane layout: M = lane%16; lanes<16 hold K {kb..kb+15},
// lanes>=16 hold K {kb+16..kb+31}.  Two ds_load_b128 per fragment.
__device__ __forceinline__ v16bf load_A_lds(const unsigned short* lds, int stride,
                                            int kb, int lane) {
  const unsigned short* p = lds + (lane & 15) * stride + kb + (lane >> 4) * 16;
  BF16x16 f;
  uint4 q0 = *(const uint4*)(p);
  uint4 q1 = *(const uint4*)(p + 8);
  f.u[0] = q0.x; f.u[1] = q0.y; f.u[2] = q0.z; f.u[3] = q0.w;
  f.u[4] = q1.x; f.u[5] = q1.y; f.u[6] = q1.z; f.u[7] = q1.w;
  return f.v;
}

// B fragment where B[k][n] = W[n][k] (row-major W, ld=ldw).
// Lane column n = nrow_base + lane%16; K = kb + (lane>=16?16:0) + 0..15
// -> 16 contiguous fp32 per lane.  All four b128 loads are issued with
// independent destinations before any packing (one clause, one wait).
__device__ __forceinline__ v16bf load_B_rows(const float* __restrict__ W, int ldw,
                                             int nrow_base, int kb, int lane) {
  const float* p = W + (size_t)(nrow_base + (lane & 15)) * ldw + kb + (lane >> 4) * 16;
  float4 q0 = *(const float4*)(p);
  float4 q1 = *(const float4*)(p + 4);
  float4 q2 = *(const float4*)(p + 8);
  float4 q3 = *(const float4*)(p + 12);
  BF16x16 f;
  f.u[0] = pack_bf16(q0.x, q0.y);
  f.u[1] = pack_bf16(q0.z, q0.w);
  f.u[2] = pack_bf16(q1.x, q1.y);
  f.u[3] = pack_bf16(q1.z, q1.w);
  f.u[4] = pack_bf16(q2.x, q2.y);
  f.u[5] = pack_bf16(q2.z, q2.w);
  f.u[6] = pack_bf16(q3.x, q3.y);
  f.u[7] = pack_bf16(q3.z, q3.w);
  return f.v;
}

// B fragment for the staged V chunk: 8 conflict-free ds_load_b32.
// Pair row index = (lane>=16 ? 8 : 0) + i; column = col_base + lane%16.
__device__ __forceinline__ v16bf load_B_vstage(const unsigned int* vs,
                                               int col_base, int lane) {
  const unsigned int* p = vs + (lane >> 4) * 8 * VSTR + col_base + (lane & 15);
  BF16x16 f;
#pragma unroll
  for (int i = 0; i < 8; ++i) f.u[i] = p[i * VSTR];
  return f.v;
}

// ---------------------------------------------------------------------------
// Flash split-K partial: grid (16 M-tiles, nsplit).  Each block walks its
// slot range in 128-slot tiles, keeping online softmax state and a 16x1024
// fp32 output accumulator spread across 8 waves (16x128 each).
// ---------------------------------------------------------------------------
__global__ __launch_bounds__(256) void flash_partial(
    const float* __restrict__ x, const float* __restrict__ W,
    const float* __restrict__ bias, const float* __restrict__ Vm,
    int n_len, float* __restrict__ part) {
  __shared__ unsigned short xb[MT * 1032];   // x tile, bf16 (~33 KB)
  __shared__ unsigned short pbuf[MT * 136];  // softmax probs tile, bf16
  __shared__ unsigned int   vstage[16 * VSTR]; // pair-packed V chunk (~66 KB)
  __shared__ float wmax[8][16];
  __shared__ float wsum[8][16];
  __shared__ float mrow[16], srow[16], mnew[16], scl[16];

  const int tid  = threadIdx.x;
  const int lane = tid & 31;
  const int wave = tid >> 5;
  const int rbase = (lane >> 4) * 8;          // row base for C/D fragments
  const int mbase = blockIdx.x * MT;
  const int nbase = blockIdx.y * n_len;
  const int hb    = wave * 128;               // this wave's output column slice

  stage16x1024(x + (size_t)mbase * HD, xb, tid);
  if (tid < 16) { mrow[tid] = -__builtin_inff(); srow[tid] = 0.0f; }

  v8f zero = {0.f, 0.f, 0.f, 0.f, 0.f, 0.f, 0.f, 0.f};
  v8f acc[8];
#pragma unroll
  for (int j = 0; j < 8; ++j) acc[j] = zero;

  __syncthreads();

  for (int nt = nbase; nt < nbase + n_len; nt += NT) {
    // ---- logits: this wave's 16 slots, S = x @ W^T over K=1024 ----
    v8f sv = zero;
    const int slot0 = nt + wave * 16;
#pragma unroll 4
    for (int kb = 0; kb < HD; kb += 32) {
      v16bf a = load_A_lds(xb, 1032, kb, lane);
      v16bf b = load_B_rows(W, HD, slot0, kb, lane);
      sv = __builtin_amdgcn_wmma_f32_16x16x32_bf16(false, a, false, b,
                                                   (short)0, sv, false, false);
    }
    const float bcol = bias[slot0 + (lane & 15)];
    float s[8];
#pragma unroll
    for (int r = 0; r < 8; ++r) s[r] = sv[r] + bcol;

    // ---- wave-local row max over 16 slots (within each 16-lane half) ----
#pragma unroll
    for (int r = 0; r < 8; ++r) {
      float v = s[r];
      v = fmaxf(v, __shfl_xor(v, 1, 32));
      v = fmaxf(v, __shfl_xor(v, 2, 32));
      v = fmaxf(v, __shfl_xor(v, 4, 32));
      v = fmaxf(v, __shfl_xor(v, 8, 32));
      if ((lane & 15) == 0) wmax[wave][rbase + r] = v;
    }
    __syncthreads();
    if (tid < 16) {
      float mo = mrow[tid];
      float mx = mo;
#pragma unroll
      for (int w = 0; w < 8; ++w) mx = fmaxf(mx, wmax[w][tid]);
      mnew[tid] = mx;
      scl[tid]  = __expf(mo - mx);
      mrow[tid] = mx;
    }
    __syncthreads();

    // ---- p = exp(s - m_new); rescale O; partial row sums; stage P ----
#pragma unroll
    for (int r = 0; r < 8; ++r) {
      const float mn = mnew[rbase + r];
      const float sc = scl[rbase + r];
      float p = __expf(s[r] - mn);
#pragma unroll
      for (int j = 0; j < 8; ++j) acc[j][r] *= sc;
      float v = p;
      v += __shfl_xor(v, 1, 32);
      v += __shfl_xor(v, 2, 32);
      v += __shfl_xor(v, 4, 32);
      v += __shfl_xor(v, 8, 32);
      if ((lane & 15) == 0) wsum[wave][rbase + r] = v;
      pbuf[(rbase + r) * 136 + wave * 16 + (lane & 15)] =
          (unsigned short)(__float_as_uint(p) >> 16);
    }
    __syncthreads();
    if (tid < 16) {
      float d = srow[tid] * scl[tid];
#pragma unroll
      for (int w = 0; w < 8; ++w) d += wsum[w][tid];
      srow[tid] = d;
    }

    // ---- O += P @ V: stage each 32-slot chunk of V through LDS ----
#pragma unroll
    for (int kk = 0; kk < NT; kk += 32) {
      stageV(Vm, nt + kk, vstage, tid);
      __syncthreads();
      v16bf a = load_A_lds(pbuf, 136, kk, lane);
#pragma unroll
      for (int j = 0; j < 8; ++j) {
        v16bf b = load_B_vstage(vstage, hb + j * 16, lane);
        acc[j] = __builtin_amdgcn_wmma_f32_16x16x32_bf16(false, a, false, b,
                                                         (short)0, acc[j],
                                                         false, false);
      }
      __syncthreads();   // vstage reuse; final one also fences pbuf/wmax
    }
  }

  // ---- write partial (O, m, s) ----
  float* po = part + (size_t)(blockIdx.x * gridDim.y + blockIdx.y) * PART_STRIDE;
#pragma unroll
  for (int j = 0; j < 8; ++j)
#pragma unroll
    for (int r = 0; r < 8; ++r)
      po[(rbase + r) * HD + hb + j * 16 + (lane & 15)] = acc[j][r];
  if (tid < 16) { po[16384 + tid] = mrow[tid]; po[16400 + tid] = srow[tid]; }
}

// ---------------------------------------------------------------------------
// Combine split-K flash partials:  out = sum_i O_i * exp(m_i - m) / s
// grid.x = 16 M-tiles.
// ---------------------------------------------------------------------------
__global__ __launch_bounds__(256) void flash_reduce(
    const float* __restrict__ part, int nsplit, float* __restrict__ out) {
  const int tid = threadIdx.x;
  const int mt  = blockIdx.x;
  const float* base = part + (size_t)mt * nsplit * PART_STRIDE;
  __shared__ float mg[16], dn[16], wgt[32][16];

  if (tid < 16) {
    float m = -__builtin_inff();
    for (int s = 0; s < nsplit; ++s)
      m = fmaxf(m, base[(size_t)s * PART_STRIDE + 16384 + tid]);
    mg[tid] = m;
  }
  __syncthreads();
  for (int i = tid; i < nsplit * 16; i += 256) {
    int s = i >> 4, r = i & 15;
    wgt[s][r] = __expf(base[(size_t)s * PART_STRIDE + 16384 + r] - mg[r]);
  }
  __syncthreads();
  if (tid < 16) {
    float d = 0.0f;
    for (int s = 0; s < nsplit; ++s)
      d += base[(size_t)s * PART_STRIDE + 16400 + tid] * wgt[s][tid];
    dn[tid] = 1.0f / d;
  }
  __syncthreads();

  for (int idx = tid; idx < 16384; idx += 256) {
    int row = idx >> 10;
    float a = 0.0f;
    for (int s = 0; s < nsplit; ++s)
      a += base[(size_t)s * PART_STRIDE + idx] * wgt[s][row];
    out[(size_t)mt * 16384 + idx] = a * dn[row];
  }
}

// ---------------------------------------------------------------------------
// out = [episodic, x] @ Wc^T + bc.  grid (16 M-tiles, 8 column chunks of 128).
// ---------------------------------------------------------------------------
__global__ __launch_bounds__(256) void consolidate(
    const float* __restrict__ ep, const float* __restrict__ x,
    const float* __restrict__ Wc, const float* __restrict__ bc,
    float* __restrict__ out) {
  __shared__ unsigned short ab[MT * 1032];
  const int tid  = threadIdx.x;
  const int lane = tid & 31;
  const int wave = tid >> 5;
  const int rbase = (lane >> 4) * 8;
  const int mbase = blockIdx.x * MT;
  const int col0  = blockIdx.y * 128 + wave * 16;

  v8f acc = {0.f, 0.f, 0.f, 0.f, 0.f, 0.f, 0.f, 0.f};

  stage16x1024(ep + (size_t)mbase * HD, ab, tid);
  __syncthreads();
#pragma unroll 4
  for (int kb = 0; kb < HD; kb += 32) {
    v16bf a = load_A_lds(ab, 1032, kb, lane);
    v16bf b = load_B_rows(Wc, 2 * HD, col0, kb, lane);
    acc = __builtin_amdgcn_wmma_f32_16x16x32_bf16(false, a, false, b,
                                                  (short)0, acc, false, false);
  }
  __syncthreads();
  stage16x1024(x + (size_t)mbase * HD, ab, tid);
  __syncthreads();
#pragma unroll 4
  for (int kb = 0; kb < HD; kb += 32) {
    v16bf a = load_A_lds(ab, 1032, kb, lane);
    v16bf b = load_B_rows(Wc, 2 * HD, col0, HD + kb, lane);
    acc = __builtin_amdgcn_wmma_f32_16x16x32_bf16(false, a, false, b,
                                                  (short)0, acc, false, false);
  }

  const int col  = col0 + (lane & 15);
  const float bb = bc[col];
#pragma unroll
  for (int r = 0; r < 8; ++r)
    out[(size_t)(mbase + rbase + r) * HD + col] = acc[r] + bb;
}

// ---------------------------------------------------------------------------
extern "C" void kernel_launch(void* const* d_in, const int* in_sizes, int n_in,
                              void* d_out, int out_size, void* d_ws, size_t ws_size,
                              hipStream_t stream) {
  (void)in_sizes; (void)n_in; (void)out_size; (void)ws_size;
  const float* x    = (const float*)d_in[0];
  const float* We   = (const float*)d_in[1];
  const float* be   = (const float*)d_in[2];
  const float* epm  = (const float*)d_in[3];
  const float* Ws   = (const float*)d_in[4];
  const float* bs   = (const float*)d_in[5];
  const float* semm = (const float*)d_in[6];
  const float* Wc   = (const float*)d_in[7];
  const float* bc   = (const float*)d_in[8];
  float* out = (float*)d_out;   // [256*1024] out, then [256*1024] semantic

  const int NS_E = 16;   // slot splits for episodic (65536 -> 4096 each)
  const int NS_S = 32;   // slot splits for semantic (131072 -> 4096 each)

  float* part_e = (float*)d_ws;                               // 16*NS_E partials
  float* part_s = part_e + (size_t)16 * NS_E * PART_STRIDE;   // 16*NS_S partials
  float* ep_out = part_s + (size_t)16 * NS_S * PART_STRIDE;   // [256][1024]

  flash_partial<<<dim3(16, NS_E), 256, 0, stream>>>(x, We, be, epm,
                                                    65536 / NS_E, part_e);
  flash_partial<<<dim3(16, NS_S), 256, 0, stream>>>(x, Ws, bs, semm,
                                                    131072 / NS_S, part_s);
  flash_reduce<<<dim3(16), 256, 0, stream>>>(part_e, NS_E, ep_out);
  flash_reduce<<<dim3(16), 256, 0, stream>>>(part_s, NS_S, out + 256 * 1024);
  consolidate<<<dim3(16, 8), 256, 0, stream>>>(ep_out, x, Wc, bc, out);
}
